// BiLSTMNet_30133490549691
// MI455X (gfx1250) — matrix operand order repaired and checked by reading, not compiled, not run on hardware
//
#include <hip/hip_runtime.h>
#include <cstddef>

typedef __attribute__((ext_vector_type(16))) _Float16 v16h;
typedef __attribute__((ext_vector_type(8))) float v8f;

union F16Frag { v16h v; unsigned int u[8]; };

// A fragment: 16x32 f16 from row-major [rows][ld] (ld in f16 elems, 16B-aligned rows).
// Lane L: M = L%16 ; K = (L<16 ? 0..15 : 16..31)
__device__ __forceinline__ v16h load_a_frag(const _Float16* base, int row0, int ld, int k0, int lane) {
  const _Float16* p = base + (size_t)(row0 + (lane & 15)) * (size_t)ld + (size_t)(k0 + ((lane & 16) ? 16 : 0));
  F16Frag f;
#pragma unroll
  for (int i = 0; i < 8; ++i) {
    f.u[i]     = *(const unsigned int*)(p + 2 * i);
  }
  return f.v;
}

// B fragment: 32(K)x16(N) f16 from weights stored [n][ld] (n-major, k contiguous).
// Lane L: N = L%16 ; K = (L<16 ? 0..15 : 16..31), 2 K per VGPR.
__device__ __forceinline__ v16h load_b_frag(const _Float16* base, int col0, int ld, int k0, int lane) {
  const _Float16* p = base + (size_t)(col0 + (lane & 15)) * (size_t)ld + (size_t)(k0 + ((lane & 16) ? 16 : 0));
  F16Frag f;
#pragma unroll
  for (int i = 0; i < 8; ++i) f.u[i] = *(const unsigned int*)(p + 2 * i);
  return f.v;
}

#define WMMA_F32_F16(a, b, c) \
  __builtin_amdgcn_wmma_f32_16x16x32_f16(false, (a), false, (b), (short)0, (c), false, false)

__device__ __forceinline__ float sigmoidf_(float x) { return 1.0f / (1.0f + __expf(-x)); }

// ---------------------------------------------------------------------------
// Weight prep: f32 [Nsrc][Ksrc] -> f16 [Npad][Kp].
// N map: gatepad ? (4 gates of 200 -> 4 x 208) : identity (pad rows >= Nsrc).
// K map (uniform for all weights): k' = d*224 + j  <->  src k = d*200 + j (j<200).
// ---------------------------------------------------------------------------
__global__ void prep_w_kernel(const float* __restrict__ src, _Float16* __restrict__ dst,
                              int Kp, int Ksrc, int Npad, int gatepad, int Nsrc) {
  int i = blockIdx.x * blockDim.x + threadIdx.x;
  if (i >= Npad * Kp) return;
  int n = i / Kp, k = i - n * Kp;
  int srow;
  if (gatepad) { int g = n / 208, j = n - g * 208; srow = (j < 200) ? (g * 200 + j) : -1; }
  else srow = (n < Nsrc) ? n : -1;
  int d = k / 224, j = k - d * 224;
  int scol = (j < 200) ? (d * 200 + j) : -1;
  float v = (srow >= 0 && scol >= 0) ? src[(size_t)srow * Ksrc + scol] : 0.0f;
  dst[i] = (_Float16)v;
}

__global__ void prep_b_kernel(const float* __restrict__ src, float* __restrict__ dst,
                              int Npad, int gatepad, int Nsrc) {
  int n = blockIdx.x * blockDim.x + threadIdx.x;
  if (n >= Npad) return;
  int srow;
  if (gatepad) { int g = n / 208, j = n - g * 208; srow = (j < 200) ? (g * 200 + j) : -1; }
  else srow = (n < Nsrc) ? n : -1;
  dst[n] = (srow >= 0) ? src[srow] : 0.0f;
}

// ---------------------------------------------------------------------------
// Zero an f16 buffer (count must be a multiple of 8).
// ---------------------------------------------------------------------------
__global__ void zero_f16_kernel(_Float16* __restrict__ p, long long n8) {
  long long i = (long long)blockIdx.x * blockDim.x + threadIdx.x;
  if (i < n8) ((float4*)p)[i] = make_float4(0.f, 0.f, 0.f, 0.f);
}

// ---------------------------------------------------------------------------
// Embedding gather -> padded f16 x [65536][224]; row m = t*128+b.
// ---------------------------------------------------------------------------
__global__ __launch_bounds__(256) void embed_kernel(const int* __restrict__ tokens,
                                                    const float* __restrict__ emb,
                                                    _Float16* __restrict__ xemb) {
  int idx = blockIdx.x * blockDim.x + threadIdx.x;  // 65536 * 28
  int row = idx / 28;
  int c8 = (idx - row * 28) * 8;
  _Float16* dp = xemb + (size_t)row * 224 + c8;
  if (c8 >= 200) {
#pragma unroll
    for (int i = 0; i < 8; ++i) dp[i] = (_Float16)0.0f;
  } else {
    int t = row >> 7, b = row & 127;
    const float* ep = emb + (size_t)tokens[b * 512 + t] * 200 + c8;
#pragma unroll
    for (int i = 0; i < 8; ++i) dp[i] = (_Float16)ep[i];
  }
}

// ---------------------------------------------------------------------------
// Conf gather -> padded f16 mlp_in [32768][896] from padded h1 [65536][448].
// ---------------------------------------------------------------------------
__global__ __launch_bounds__(256) void mlp_gather_kernel(const int* __restrict__ confs,
                                                         const _Float16* __restrict__ h1,
                                                         _Float16* __restrict__ mlpin) {
  int idx = blockIdx.x * blockDim.x + threadIdx.x;  // 32768 * 112
  int row = idx / 112;
  int c8 = (idx - row * 112) * 8;
  int blk = (c8 >= 448) ? 1 : 0;
  int j = c8 - blk * 448;
  int conf = confs[row * 2 + blk];
  float4* dp = (float4*)(mlpin + (size_t)row * 896 + c8);
  if (conf < 0) {
    *dp = make_float4(0.f, 0.f, 0.f, 0.f);
  } else {
    int b = row >> 8;
    *dp = *(const float4*)(h1 + ((size_t)conf * 128 + b) * 448 + j);
  }
}

// ---------------------------------------------------------------------------
// WMMA GEMM, no LDS, no guards, software-pipelined.
// out[m][n] = act( sum_k A[m][k]*W[n][k] + bias[n] )
// Block = 8 waves over a 64(M) x 64(N) tile; each wave owns two 16x16 C tiles
// sharing one A fragment. Next chunk's fragments are prefetched while the
// current chunk's wmma pair executes.
// ---------------------------------------------------------------------------
__global__ __launch_bounds__(256) void wmma_gemm_kernel(
    const _Float16* __restrict__ A, const _Float16* __restrict__ W, int ld, int nkc,
    const float* __restrict__ bias, float* __restrict__ out, int out_ld, int Nstore, int act) {
  const int lane = threadIdx.x & 31;
  const int wave = threadIdx.x >> 5;
  const int arow = blockIdx.x * 64 + (wave & 3) * 16;
  const int bcol = blockIdx.y * 64 + (wave >> 2) * 32;  // two N tiles: bcol, bcol+16

  v8f acc0 = {}, acc1 = {};
  v16h a  = load_a_frag(A, arow, ld, 0, lane);
  v16h b0 = load_b_frag(W, bcol, ld, 0, lane);
  v16h b1 = load_b_frag(W, bcol + 16, ld, 0, lane);
  for (int kc = 0; kc + 1 < nkc; ++kc) {
    const int k1 = (kc + 1) * 32;
    v16h an  = load_a_frag(A, arow, ld, k1, lane);
    v16h b0n = load_b_frag(W, bcol, ld, k1, lane);
    v16h b1n = load_b_frag(W, bcol + 16, ld, k1, lane);
    acc0 = WMMA_F32_F16(a, b0, acc0);
    acc1 = WMMA_F32_F16(a, b1, acc1);
    a = an; b0 = b0n; b1 = b1n;
  }
  acc0 = WMMA_F32_F16(a, b0, acc0);
  acc1 = WMMA_F32_F16(a, b1, acc1);

  const int mrow = arow + ((lane & 16) ? 8 : 0);
  const int n0 = bcol + (lane & 15);
  if (n0 < Nstore) {
    const float bv = bias[n0];
#pragma unroll
    for (int r = 0; r < 8; ++r) {
      float v = acc0[r] + bv;
      if (act) v = tanhf(v);
      out[(size_t)(mrow + r) * out_ld + n0] = v;
    }
  }
  const int n1 = n0 + 16;
  if (n1 < Nstore) {
    const float bv = bias[n1];
#pragma unroll
    for (int r = 0; r < 8; ++r) {
      float v = acc1[r] + bv;
      if (act) v = tanhf(v);
      out[(size_t)(mrow + r) * out_ld + n1] = v;
    }
  }
}

// ---------------------------------------------------------------------------
// Persistent LSTM recurrence. grid = 2 (block 0: fwd, block 1: bwd),
// block = 512 threads (16 waves). LDS: c f32 [128][208], h f16 double-buffered
// [2][128][224]. gates = xg[t] + h @ Whh^T (WMMA, pipelined) fused pointwise.
// Output h written as padded f16 [t*128+m][448] at col dir*224 + j.
// ---------------------------------------------------------------------------
__global__ __launch_bounds__(512) void lstm_rec_kernel(
    const float* __restrict__ xgF, const float* __restrict__ xgB,
    const _Float16* __restrict__ WhhF, const _Float16* __restrict__ WhhB,
    _Float16* __restrict__ h_out) {
  extern __shared__ char smem[];
  float* cst = (float*)smem;                          // 128*208 f32
  _Float16* hb0 = (_Float16*)(smem + 128 * 208 * 4);  // 128*224 f16
  _Float16* hb1 = hb0 + 128 * 224;

  const int dir = blockIdx.x;
  const float* xg = dir ? xgB : xgF;
  const _Float16* Whh = dir ? WhhB : WhhF;
  const int tid = threadIdx.x;
  const int lane = tid & 31;
  const int wave = tid >> 5;

  // zero c and both h buffers (221184 bytes = 55296 dwords)
  for (int i = tid; i < 55296; i += 512) ((unsigned int*)smem)[i] = 0u;
  __syncthreads();

  for (int s = 0; s < 512; ++s) {
    const int t = dir ? (511 - s) : s;
    _Float16* hcur = (s & 1) ? hb1 : hb0;
    _Float16* hnxt = (s & 1) ? hb0 : hb1;
    const float* xgt = xg + (size_t)t * 128 * 832;

    // hint-prefetch next step's xg slab (512 threads x 208-float stride == 128*832)
    if (s + 1 < 512) {
      const float* nx = xg + (size_t)(dir ? (510 - s) : (s + 1)) * 128 * 832;
      __builtin_prefetch(nx + (size_t)tid * 208, 0, 1);
    }

    for (int u = wave; u < 104; u += 16) {
      const int mi = u & 7;    // batch tile, 8 x 16
      const int jt = u >> 3;   // hidden-col tile, 13 x 16 (208 padded)
      const int nlo = jt * 16 + (lane & 15);
      const int mbase = mi * 16 + ((lane & 16) ? 8 : 0);
      const int nb = jt * 16;

      v8f acc0 = {}, acc1 = {}, acc2 = {}, acc3 = {};
#pragma unroll
      for (int r = 0; r < 8; ++r) {  // accumulators start from xg (bias folded in)
        const float* xr = xgt + (size_t)(mbase + r) * 832;
        acc0[r] = xr[nlo];
        acc1[r] = xr[208 + nlo];
        acc2[r] = xr[416 + nlo];
        acc3[r] = xr[624 + nlo];
      }

      // pipelined K loop: prefetch chunk kc+1 while chunk kc's wmma execute
      v16h a_c  = load_a_frag(hcur, mi * 16, 224, 0, lane);
      v16h b0_c = load_b_frag(Whh, nb, 224, 0, lane);
      v16h b1_c = load_b_frag(Whh, 208 + nb, 224, 0, lane);
      v16h b2_c = load_b_frag(Whh, 416 + nb, 224, 0, lane);
      v16h b3_c = load_b_frag(Whh, 624 + nb, 224, 0, lane);
#pragma unroll
      for (int kc = 0; kc < 7; ++kc) {
        v16h a_n = a_c, b0_n = b0_c, b1_n = b1_c, b2_n = b2_c, b3_n = b3_c;
        if (kc < 6) {
          const int k1 = (kc + 1) * 32;
          a_n  = load_a_frag(hcur, mi * 16, 224, k1, lane);
          b0_n = load_b_frag(Whh, nb, 224, k1, lane);
          b1_n = load_b_frag(Whh, 208 + nb, 224, k1, lane);
          b2_n = load_b_frag(Whh, 416 + nb, 224, k1, lane);
          b3_n = load_b_frag(Whh, 624 + nb, 224, k1, lane);
        }
        acc0 = WMMA_F32_F16(a_c, b0_c, acc0);
        acc1 = WMMA_F32_F16(a_c, b1_c, acc1);
        acc2 = WMMA_F32_F16(a_c, b2_c, acc2);
        acc3 = WMMA_F32_F16(a_c, b3_c, acc3);
        a_c = a_n; b0_c = b0_n; b1_c = b1_n; b2_c = b2_n; b3_c = b3_n;
      }

      // fused pointwise: c' = sig(f)*c + sig(i)*tanh(g); h = sig(o)*tanh(c')
      const int j = jt * 16 + (lane & 15);
#pragma unroll
      for (int r = 0; r < 8; ++r) {
        const int m = mbase + r;
        float iv = sigmoidf_(acc0[r]);
        float fv = sigmoidf_(acc1[r]);
        float gv = tanhf(acc2[r]);
        float ov = sigmoidf_(acc3[r]);
        float cn = fv * cst[m * 208 + j] + iv * gv;
        cst[m * 208 + j] = cn;
        float hv = ov * tanhf(cn);
        hnxt[m * 224 + j] = (_Float16)hv;  // pad cols compute exactly 0
        h_out[((size_t)t * 128 + m) * 448 + dir * 224 + j] = (_Float16)hv;
      }
    }
    __syncthreads();  // h double-buffer swap point
  }
}

// ---------------------------------------------------------------------------
// Final head: out = softmax(hmid @ w2^T + bw2) over 4 classes. One thread/row.
// ---------------------------------------------------------------------------
__global__ __launch_bounds__(256) void final_head_kernel(
    const float* __restrict__ hmid, const float* __restrict__ w2,
    const float* __restrict__ bw2, float* __restrict__ out) {
  const int row = blockIdx.x * 256 + threadIdx.x;
  const float* hp = hmid + (size_t)row * 400;
  float s0 = bw2[0], s1 = bw2[1], s2 = bw2[2], s3 = bw2[3];
  for (int k = 0; k < 400; ++k) {
    float h = hp[k];
    s0 += h * w2[k];
    s1 += h * w2[400 + k];
    s2 += h * w2[800 + k];
    s3 += h * w2[1200 + k];
  }
  float mx = fmaxf(fmaxf(s0, s1), fmaxf(s2, s3));
  float e0 = __expf(s0 - mx), e1 = __expf(s1 - mx), e2 = __expf(s2 - mx), e3 = __expf(s3 - mx);
  float inv = 1.0f / (e0 + e1 + e2 + e3);
  out[(size_t)row * 4 + 0] = e0 * inv;
  out[(size_t)row * 4 + 1] = e1 * inv;
  out[(size_t)row * 4 + 2] = e2 * inv;
  out[(size_t)row * 4 + 3] = e3 * inv;
}

// ---------------------------------------------------------------------------
extern "C" void kernel_launch(void* const* d_in, const int* in_sizes, int n_in,
                              void* d_out, int out_size, void* d_ws, size_t ws_size,
                              hipStream_t stream) {
  (void)in_sizes; (void)n_in; (void)out_size; (void)ws_size;
  const float* emb   = (const float*)d_in[0];
  const float* Wih0f = (const float*)d_in[1];
  const float* Whh0f = (const float*)d_in[2];
  const float* b0f   = (const float*)d_in[3];
  const float* Wih0b = (const float*)d_in[4];
  const float* Whh0b = (const float*)d_in[5];
  const float* b0b   = (const float*)d_in[6];
  const float* Wih1f = (const float*)d_in[7];
  const float* Whh1f = (const float*)d_in[8];
  const float* b1f   = (const float*)d_in[9];
  const float* Wih1b = (const float*)d_in[10];
  const float* Whh1b = (const float*)d_in[11];
  const float* b1b   = (const float*)d_in[12];
  const float* w1    = (const float*)d_in[13];
  const float* bw1   = (const float*)d_in[14];
  const float* w2    = (const float*)d_in[15];
  const float* bw2   = (const float*)d_in[16];
  const int* tokens  = (const int*)d_in[17];
  const int* confs   = (const int*)d_in[18];
  float* outp = (float*)d_out;

  char* ws = (char*)d_ws;
  size_t off = 0;
  auto alloc = [&](size_t bytes) -> void* {
    void* p = ws + off;
    off = (off + bytes + 255) & ~(size_t)255;
    return p;
  };

  _Float16* wA0f = (_Float16*)alloc((size_t)832 * 224 * 2);
  _Float16* wA0b = (_Float16*)alloc((size_t)832 * 224 * 2);
  _Float16* wH0f = (_Float16*)alloc((size_t)832 * 224 * 2);
  _Float16* wH0b = (_Float16*)alloc((size_t)832 * 224 * 2);
  _Float16* wA1f = (_Float16*)alloc((size_t)832 * 448 * 2);
  _Float16* wA1b = (_Float16*)alloc((size_t)832 * 448 * 2);
  _Float16* wH1f = (_Float16*)alloc((size_t)832 * 224 * 2);
  _Float16* wH1b = (_Float16*)alloc((size_t)832 * 224 * 2);
  _Float16* wW1  = (_Float16*)alloc((size_t)448 * 896 * 2);
  float* pb0f = (float*)alloc(832 * 4);
  float* pb0b = (float*)alloc(832 * 4);
  float* pb1f = (float*)alloc(832 * 4);
  float* pb1b = (float*)alloc(832 * 4);
  float* pbw1 = (float*)alloc(448 * 4);
  _Float16* xemb  = (_Float16*)alloc((size_t)65536 * 224 * 2);
  _Float16* h0f16 = (_Float16*)alloc((size_t)65536 * 448 * 2);
  _Float16* h1f16 = (_Float16*)alloc((size_t)65536 * 448 * 2);
  _Float16* mlpin = (_Float16*)alloc((size_t)32768 * 896 * 2);
  float* xgF  = (float*)alloc((size_t)512 * 128 * 832 * 4);
  float* xgB  = (float*)alloc((size_t)512 * 128 * 832 * 4);
  float* hmid = (float*)alloc((size_t)32768 * 400 * 4);

  // ---- weight / bias prep into padded f16 layouts ----
  auto prepW = [&](const float* s, _Float16* d, int Kp, int Ksrc, int Npad, int gp, int Nsrc) {
    int n = Npad * Kp;
    prep_w_kernel<<<(n + 255) / 256, 256, 0, stream>>>(s, d, Kp, Ksrc, Npad, gp, Nsrc);
  };
  prepW(Wih0f, wA0f, 224, 200, 832, 1, 800);
  prepW(Wih0b, wA0b, 224, 200, 832, 1, 800);
  prepW(Whh0f, wH0f, 224, 200, 832, 1, 800);
  prepW(Whh0b, wH0b, 224, 200, 832, 1, 800);
  prepW(Wih1f, wA1f, 448, 400, 832, 1, 800);
  prepW(Wih1b, wA1b, 448, 400, 832, 1, 800);
  prepW(Whh1f, wH1f, 224, 200, 832, 1, 800);
  prepW(Whh1b, wH1b, 224, 200, 832, 1, 800);
  prepW(w1,    wW1,  896, 800, 448, 0, 400);
  prep_b_kernel<<<4, 256, 0, stream>>>(b0f, pb0f, 832, 1, 800);
  prep_b_kernel<<<4, 256, 0, stream>>>(b0b, pb0b, 832, 1, 800);
  prep_b_kernel<<<4, 256, 0, stream>>>(b1f, pb1f, 832, 1, 800);
  prep_b_kernel<<<4, 256, 0, stream>>>(b1b, pb1b, 832, 1, 800);
  prep_b_kernel<<<2, 256, 0, stream>>>(bw1, pbw1, 448, 0, 400);

  // ---- activation prep ----
  embed_kernel<<<(65536 * 28) / 256, 256, 0, stream>>>(tokens, emb, xemb);
  {
    long long n8 = (long long)65536 * 448 / 8;
    int blocks = (int)((n8 + 255) / 256);
    zero_f16_kernel<<<blocks, 256, 0, stream>>>(h0f16, n8);
    zero_f16_kernel<<<blocks, 256, 0, stream>>>(h1f16, n8);
  }

  const size_t recLds = (size_t)128 * 208 * 4 + 2 * (size_t)128 * 224 * 2;  // 221184 B

  // ---- layer 0 ----
  dim3 g0(65536 / 64, 832 / 64);
  wmma_gemm_kernel<<<g0, 256, 0, stream>>>(xemb, wA0f, 224, 7, pb0f, xgF, 832, 832, 0);
  wmma_gemm_kernel<<<g0, 256, 0, stream>>>(xemb, wA0b, 224, 7, pb0b, xgB, 832, 832, 0);
  lstm_rec_kernel<<<2, 512, recLds, stream>>>(xgF, xgB, wH0f, wH0b, h0f16);

  // ---- layer 1 ----
  wmma_gemm_kernel<<<g0, 256, 0, stream>>>(h0f16, wA1f, 448, 14, pb1f, xgF, 832, 832, 0);
  wmma_gemm_kernel<<<g0, 256, 0, stream>>>(h0f16, wA1b, 448, 14, pb1b, xgB, 832, 832, 0);
  lstm_rec_kernel<<<2, 512, recLds, stream>>>(xgF, xgB, wH1f, wH1b, h1f16);

  // ---- MLP ----
  mlp_gather_kernel<<<(32768 * 112) / 256, 256, 0, stream>>>(confs, h1f16, mlpin);
  dim3 gm(32768 / 64, 7);
  wmma_gemm_kernel<<<gm, 256, 0, stream>>>(mlpin, wW1, 896, 28, pbw1, hmid, 400, 400, 1);

  // ---- head ----
  final_head_kernel<<<32768 / 256, 256, 0, stream>>>(hmid, w2, bw2, outp);
}
